// MultiHeadAttention_3530463117935
// MI455X (gfx1250) — compile-verified, matrix operand order checked by reading, not run on hardware
//
#include <hip/hip_runtime.h>
#include <hip/hip_bf16.h>

typedef unsigned short ushort_t;
typedef unsigned int   uint_t;
typedef __attribute__((ext_vector_type(16))) __bf16 v16bf;
typedef __attribute__((ext_vector_type(8)))  float  v8f;

#define HEADS  16
#define D_HEAD 64
#define UNITS  1024
#define IN_DIM 1024
#define B_SZ   2
#define LQ     2048
#define LK     2048
#define M_TOT  (B_SZ * LQ) /* 4096 */
#define VERY_NEG (-1e29f)

union FragU { v16bf v; uint4 q[2]; };

__device__ __forceinline__ ushort_t f2bf(float x) {
  uint_t u = __float_as_uint(x);
  u += 0x7FFFu + ((u >> 16) & 1u);   // round-to-nearest-even
  return (ushort_t)(u >> 16);
}

// 16x32 bf16 fragment load (A layout; B layout identical with N<->M).
// lane<16: row lane&15, K 0..7 & 16..23 ; lane>=16: same row, K 8..15 & 24..31.
__device__ __forceinline__ v16bf load_frag16x32(const ushort_t* base, int pitch, int lane) {
  const int r  = lane & 15;
  const int ko = ((lane >> 4) & 1) * 8;
  FragU f;
  f.q[0] = *(const uint4*)(base + r * pitch + ko);
  f.q[1] = *(const uint4*)(base + r * pitch + ko + 16);
  return f.v;
}

#define WMMA_BF16(A, Bm, C) \
  __builtin_amdgcn_wmma_f32_16x16x32_bf16(false, (A), false, (Bm), (short)0, (C), false, false)

// ---------------------------------------------------------------------------
// Pre-pass A: convert query/key/value f32 -> bf16, row-major (one-time work).
// ---------------------------------------------------------------------------
__global__ __launch_bounds__(256) void convert_x(
    const float* __restrict__ q, const float* __restrict__ k, const float* __restrict__ v,
    ushort_t* __restrict__ Xb)
{
  const float* src = (blockIdx.z == 0) ? q : (blockIdx.z == 1 ? k : v);
  ushort_t* dst = Xb + (size_t)blockIdx.z * (size_t)M_TOT * IN_DIM;
  const size_t i0 = ((size_t)blockIdx.x * 256 + threadIdx.x) * 8;
  const float4 f0 = *(const float4*)(src + i0);
  const float4 f1 = *(const float4*)(src + i0 + 4);
  uint4 o;
  o.x = (uint_t)f2bf(f0.x) | ((uint_t)f2bf(f0.y) << 16);
  o.y = (uint_t)f2bf(f0.z) | ((uint_t)f2bf(f0.w) << 16);
  o.z = (uint_t)f2bf(f1.x) | ((uint_t)f2bf(f1.y) << 16);
  o.w = (uint_t)f2bf(f1.z) | ((uint_t)f2bf(f1.w) << 16);
  *(uint4*)(dst + i0) = o;
}

// ---------------------------------------------------------------------------
// Pre-pass B: convert Wq/Wk/Wv f32 [k][n] -> bf16 transposed [n][k] (one-time).
// ---------------------------------------------------------------------------
__global__ __launch_bounds__(256) void convert_w(
    const float* __restrict__ Wq, const float* __restrict__ Wk, const float* __restrict__ Wv,
    ushort_t* __restrict__ Wt)
{
  __shared__ ushort_t T[64][72];
  const float* W = (blockIdx.z == 0) ? Wq : (blockIdx.z == 1 ? Wk : Wv);
  ushort_t* dst = Wt + (size_t)blockIdx.z * (size_t)UNITS * IN_DIM;
  const int tid = threadIdx.x;
  const int k0 = blockIdx.x * 64, n0 = blockIdx.y * 64;
  #pragma unroll
  for (int i = 0; i < 4; ++i) {
    const int idx = tid + i * 256;
    const int r  = idx >> 4;           // k-local 0..63
    const int c4 = (idx & 15) * 4;     // n-local 0..60
    const float4 f = *(const float4*)(W + (size_t)(k0 + r) * UNITS + n0 + c4);
    T[c4 + 0][r] = f2bf(f.x);
    T[c4 + 1][r] = f2bf(f.y);
    T[c4 + 2][r] = f2bf(f.z);
    T[c4 + 3][r] = f2bf(f.w);
  }
  __syncthreads();
  #pragma unroll
  for (int i = 0; i < 2; ++i) {
    const int idx = tid + i * 256;
    const int r   = idx >> 3;          // n-local 0..63
    const int seg = (idx & 7) * 8;     // k-local
    *(uint4*)(dst + (size_t)(n0 + r) * IN_DIM + k0 + seg) = *(const uint4*)&T[r][seg];
  }
}

// ---------------------------------------------------------------------------
// Kernel 1: projection GEMM C = Xb @ Wt^T + b (all bf16 operands, f32 accum).
// 128x128 block tile, 8 waves x (32x64), k-step 64, double-buffered LDS,
// one barrier per k-step. blockIdx.z selects {Q,K,V}; Q pre-scaled by 0.125.
// Q,K out: [B][H][L][64]; V out transposed: [B][H][64][L].
// ---------------------------------------------------------------------------
__global__ __launch_bounds__(256) void proj_kernel(
    const ushort_t* __restrict__ Xb, const ushort_t* __restrict__ Wt,
    const float* __restrict__ bq, const float* __restrict__ bk, const float* __restrict__ bv,
    ushort_t* __restrict__ Qp, ushort_t* __restrict__ Kp, ushort_t* __restrict__ Vt)
{
  __shared__ ushort_t Al[2][128][72];   // [buf][m][k], 144B rows (16B aligned)
  __shared__ ushort_t Bl[2][128][72];   // [buf][n][k]

  const int tid  = threadIdx.x;
  const int lane = tid & 31;
  const int wave = tid >> 5;
  const int mi = wave >> 1;             // 0..3 -> 32-row strip
  const int ni = wave & 1;              // 0..1 -> 64-col strip
  const int mBase = blockIdx.x * 128;
  const int nBase = blockIdx.y * 128;
  const int mode  = blockIdx.z;

  const ushort_t* Xg = Xb + (size_t)mode * (size_t)M_TOT * IN_DIM;
  const ushort_t* Wg = Wt + (size_t)mode * (size_t)UNITS * IN_DIM;
  const float* bias  = (mode == 0) ? bq : (mode == 1 ? bk : bv);
  ushort_t* Out      = (mode == 0) ? Qp : (mode == 1 ? Kp : Vt);

  v8f zero = {};
  v8f acc[2][4];
  #pragma unroll
  for (int a = 0; a < 2; ++a)
    #pragma unroll
    for (int j = 0; j < 4; ++j) acc[a][j] = zero;

  const int crow = tid >> 3;            // 0..31 base staging row
  const int cseg = (tid & 7) * 8;       // k-segment (8 bf16 = 16B)

  // prologue: stage k0 = 0 into buffer 0
  #pragma unroll
  for (int i = 0; i < 4; ++i) {
    const int row = crow + 32 * i;
    *(uint4*)&Al[0][row][cseg] = *(const uint4*)(Xg + (size_t)(mBase + row) * IN_DIM + cseg);
    *(uint4*)&Bl[0][row][cseg] = *(const uint4*)(Wg + (size_t)(nBase + row) * IN_DIM + cseg);
  }

  int p = 0;
  for (int kk = 0; kk < IN_DIM / 64; ++kk) {
    __syncthreads();
    uint4 ar[4], br[4];
    const bool pf = (kk + 1) < IN_DIM / 64;
    if (pf) {                            // issue next chunk's global loads early
      const int k0 = (kk + 1) * 64;
      #pragma unroll
      for (int i = 0; i < 4; ++i) {
        const int row = crow + 32 * i;
        ar[i] = *(const uint4*)(Xg + (size_t)(mBase + row) * IN_DIM + k0 + cseg);
        br[i] = *(const uint4*)(Wg + (size_t)(nBase + row) * IN_DIM + k0 + cseg);
      }
    }
    #pragma unroll
    for (int s = 0; s < 2; ++s) {
      v16bf af0 = load_frag16x32(&Al[p][32 * mi +  0][32 * s], 72, lane);
      v16bf af1 = load_frag16x32(&Al[p][32 * mi + 16][32 * s], 72, lane);
      #pragma unroll
      for (int j = 0; j < 4; ++j) {
        v16bf bfj = load_frag16x32(&Bl[p][64 * ni + 16 * j][32 * s], 72, lane);
        acc[0][j] = WMMA_BF16(af0, bfj, acc[0][j]);
        acc[1][j] = WMMA_BF16(af1, bfj, acc[1][j]);
      }
    }
    if (pf) {                            // commit next chunk to the other buffer
      #pragma unroll
      for (int i = 0; i < 4; ++i) {
        const int row = crow + 32 * i;
        *(uint4*)&Al[p ^ 1][row][cseg] = ar[i];
        *(uint4*)&Bl[p ^ 1][row][cseg] = br[i];
      }
    }
    p ^= 1;
  }

  // epilogue
  const int rowoff = ((lane >> 4) & 1) * 8;
  #pragma unroll
  for (int a = 0; a < 2; ++a) {
    #pragma unroll
    for (int j = 0; j < 4; ++j) {
      const int col = nBase + 64 * ni + 16 * j + (lane & 15);
      const float bb = bias[col];
      const int h = col >> 6;
      const int d = col & 63;
      const int row0 = mBase + 32 * mi + 16 * a + rowoff;   // multiple of 8
      const int bi  = row0 >> 11;
      const int li0 = row0 & (LQ - 1);
      if (mode == 2) {
        // transposed V: 8 consecutive tokens, same d -> one 16B store
        float vv[8];
        #pragma unroll
        for (int r = 0; r < 8; ++r) vv[r] = acc[a][j][r] + bb;
        uint4 o;
        o.x = (uint_t)f2bf(vv[0]) | ((uint_t)f2bf(vv[1]) << 16);
        o.y = (uint_t)f2bf(vv[2]) | ((uint_t)f2bf(vv[3]) << 16);
        o.z = (uint_t)f2bf(vv[4]) | ((uint_t)f2bf(vv[5]) << 16);
        o.w = (uint_t)f2bf(vv[6]) | ((uint_t)f2bf(vv[7]) << 16);
        *(uint4*)(Out + (((size_t)bi * HEADS + h) * D_HEAD + d) * (size_t)LK + li0) = o;
      } else {
        #pragma unroll
        for (int r = 0; r < 8; ++r) {
          float val = acc[a][j][r] + bb;
          if (mode == 0) val *= 0.125f;           // fold 1/sqrt(64) into Q
          Out[(((size_t)bi * HEADS + h) * (size_t)LQ + (li0 + r)) * D_HEAD + d] = f2bf(val);
        }
      }
    }
  }
}

// ---------------------------------------------------------------------------
// Kernel 2: flash attention. 128 threads = 4 waves; wave owns 16 query rows.
// K iterated in 64-key chunks, K/V/mask double-buffered in LDS.
// ---------------------------------------------------------------------------
__global__ __launch_bounds__(128) void attn_kernel(
    const ushort_t* __restrict__ Qp, const ushort_t* __restrict__ Kp,
    const ushort_t* __restrict__ Vt, const int* __restrict__ kmask,
    float* __restrict__ out)
{
  __shared__ ushort_t Kl[2][64][72];    // [buf][key][d]
  __shared__ ushort_t Vl[2][64][72];    // [buf][d][key]
  __shared__ ushort_t Pl[4][16][72];    // per-wave P relayout buffer
  __shared__ float    Ml[2][64];        // additive key mask

  const int tid  = threadIdx.x;
  const int lane = tid & 31;
  const int wave = tid >> 5;
  const int h = blockIdx.y;
  const int b = blockIdx.z;
  const int bh = b * HEADS + h;
  const int qrow0  = blockIdx.x * 64 + wave * 16;
  const int rowoff = ((lane >> 4) & 1) * 8;

  const ushort_t* Qb = Qp + ((size_t)bh * LQ + qrow0) * D_HEAD;
  const ushort_t* Kb = Kp + (size_t)bh * LK * D_HEAD;
  const ushort_t* Vb = Vt + (size_t)bh * D_HEAD * LK;

  // Q fragments (16 rows x 64 d), loaded once from global
  v16bf qf[2];
  {
    const int r  = lane & 15;
    const int ko = ((lane >> 4) & 1) * 8;
    #pragma unroll
    for (int s = 0; s < 2; ++s) {
      FragU f;
      f.q[0] = *(const uint4*)(Qb + r * D_HEAD + 32 * s + ko);
      f.q[1] = *(const uint4*)(Qb + r * D_HEAD + 32 * s + ko + 16);
      qf[s] = f.v;
    }
  }

  v8f zero = {};
  v8f oacc[4] = {zero, zero, zero, zero};
  float mrow[8], lrow[8];
  #pragma unroll
  for (int r = 0; r < 8; ++r) { mrow[r] = -3.0e38f; lrow[r] = 0.f; }

  const int crow = tid >> 3;            // 0..15
  const int cseg = (tid & 7) * 8;

  // prologue: stage chunk 0 into buffer 0
  #pragma unroll
  for (int i = 0; i < 4; ++i) {
    const int row = crow + 16 * i;
    *(uint4*)&Kl[0][row][cseg] = *(const uint4*)(Kb + (size_t)row * D_HEAD + cseg);
    *(uint4*)&Vl[0][row][cseg] = *(const uint4*)(Vb + (size_t)row * LK + cseg);
  }
  if (tid < 64) Ml[0][tid] = (1.f - (float)kmask[b * LK + tid]) * VERY_NEG;

  int p = 0;
  for (int c = 0; c < LK / 64; ++c) {
    __syncthreads();
    uint4 kr[4], vr[4]; float mr = 0.f;
    const bool pf = (c + 1) < LK / 64;
    if (pf) {
      const int kb0 = (c + 1) * 64;
      #pragma unroll
      for (int i = 0; i < 4; ++i) {
        const int row = crow + 16 * i;
        kr[i] = *(const uint4*)(Kb + (size_t)(kb0 + row) * D_HEAD + cseg);
        vr[i] = *(const uint4*)(Vb + (size_t)row * LK + kb0 + cseg);
      }
      if (tid < 64) mr = (1.f - (float)kmask[b * LK + kb0 + tid]) * VERY_NEG;
    }

    // S = Q * K^T (16 x 64; scale folded into Q)
    v8f sacc[4] = {zero, zero, zero, zero};
    #pragma unroll
    for (int s = 0; s < 2; ++s) {
      #pragma unroll
      for (int t = 0; t < 4; ++t) {
        v16bf kbf = load_frag16x32(&Kl[p][16 * t][32 * s], 72, lane);
        sacc[t] = WMMA_BF16(qf[s], kbf, sacc[t]);
      }
    }

    // additive key mask (column = key)
    #pragma unroll
    for (int t = 0; t < 4; ++t) {
      const float ma = Ml[p][16 * t + (lane & 15)];
      #pragma unroll
      for (int r = 0; r < 8; ++r) sacc[t][r] += ma;
    }

    // online softmax: row stats live in one 16-lane half -> xor 1/2/4/8
    #pragma unroll
    for (int r = 0; r < 8; ++r) {
      float cm = fmaxf(fmaxf(sacc[0][r], sacc[1][r]), fmaxf(sacc[2][r], sacc[3][r]));
      cm = fmaxf(cm, __shfl_xor(cm, 1, 32));
      cm = fmaxf(cm, __shfl_xor(cm, 2, 32));
      cm = fmaxf(cm, __shfl_xor(cm, 4, 32));
      cm = fmaxf(cm, __shfl_xor(cm, 8, 32));
      const float nm = fmaxf(mrow[r], cm);
      const float alpha = __expf(mrow[r] - nm);
      float sum = 0.f;
      #pragma unroll
      for (int t = 0; t < 4; ++t) {
        const float pv = __expf(sacc[t][r] - nm);
        sacc[t][r] = pv;
        sum += pv;
      }
      sum += __shfl_xor(sum, 1, 32);
      sum += __shfl_xor(sum, 2, 32);
      sum += __shfl_xor(sum, 4, 32);
      sum += __shfl_xor(sum, 8, 32);
      lrow[r] = lrow[r] * alpha + sum;
      mrow[r] = nm;
      #pragma unroll
      for (int t = 0; t < 4; ++t) oacc[t][r] *= alpha;
    }

    // relayout P (C layout -> A fragment layout) through per-wave LDS buffer
    ushort_t* Pw = &Pl[wave][0][0];
    #pragma unroll
    for (int t = 0; t < 4; ++t) {
      #pragma unroll
      for (int r = 0; r < 8; ++r)
        Pw[(size_t)(r + rowoff) * 72 + 16 * t + (lane & 15)] = f2bf(sacc[t][r]);
    }

    // commit next chunk to the other buffer (disjoint from this chunk's reads)
    if (pf) {
      #pragma unroll
      for (int i = 0; i < 4; ++i) {
        const int row = crow + 16 * i;
        *(uint4*)&Kl[p ^ 1][row][cseg] = kr[i];
        *(uint4*)&Vl[p ^ 1][row][cseg] = vr[i];
      }
      if (tid < 64) Ml[p ^ 1][tid] = mr;
    }
    __syncthreads();   // orders P writes before P fragment reads

    // O += P * V
    #pragma unroll
    for (int s = 0; s < 2; ++s) {
      v16bf pa = load_frag16x32(Pw + 32 * s, 72, lane);
      #pragma unroll
      for (int t = 0; t < 4; ++t) {
        v16bf vbf = load_frag16x32(&Vl[p][16 * t][32 * s], 72, lane);
        oacc[t] = WMMA_BF16(pa, vbf, oacc[t]);
      }
    }
    p ^= 1;
  }

  float invl[8];
  #pragma unroll
  for (int r = 0; r < 8; ++r) invl[r] = 1.f / lrow[r];
  #pragma unroll
  for (int t = 0; t < 4; ++t) {
    const int dcol = h * D_HEAD + 16 * t + (lane & 15);
    #pragma unroll
    for (int r = 0; r < 8; ++r) {
      const int qi = qrow0 + rowoff + r;
      out[((size_t)b * LQ + qi) * UNITS + dcol] = oacc[t][r] * invl[r];
    }
  }
}

extern "C" void kernel_launch(void* const* d_in, const int* in_sizes, int n_in,
                              void* d_out, int out_size, void* d_ws, size_t ws_size,
                              hipStream_t stream) {
  const float* q  = (const float*)d_in[0];
  const float* k  = (const float*)d_in[1];
  const float* v  = (const float*)d_in[2];
  const int*   km = (const int*)d_in[3];
  const float* Wq = (const float*)d_in[4];
  const float* bq = (const float*)d_in[5];
  const float* Wk = (const float*)d_in[6];
  const float* bk = (const float*)d_in[7];
  const float* Wv = (const float*)d_in[8];
  const float* bv = (const float*)d_in[9];
  float* out = (float*)d_out;

  const size_t XPER = (size_t)M_TOT * IN_DIM;          // 4M bf16 per matrix
  const size_t WPER = (size_t)UNITS * IN_DIM;          // 1M bf16 per matrix
  const size_t PPER = (size_t)B_SZ * HEADS * LQ * D_HEAD; // 4M bf16 per proj

  ushort_t* Xb = (ushort_t*)d_ws;       // 3 x 8 MB
  ushort_t* Wt = Xb + 3 * XPER;         // 3 x 2 MB
  ushort_t* Qp = Wt + 3 * WPER;         // 8 MB
  ushort_t* Kp = Qp + PPER;             // 8 MB
  ushort_t* Vt = Kp + PPER;             // 8 MB

  convert_x<<<dim3((unsigned)(XPER / (256 * 8)), 1, 3), dim3(256), 0, stream>>>(q, k, v, Xb);
  convert_w<<<dim3(16, 16, 3), dim3(256), 0, stream>>>(Wq, Wk, Wv, Wt);

  dim3 g1(M_TOT / 128, UNITS / 128, 3), b1(256);
  proj_kernel<<<g1, b1, 0, stream>>>(Xb, Wt, bq, bk, bv, Qp, Kp, Vt);

  dim3 g2(LQ / 64, HEADS, B_SZ), b2(128);
  attn_kernel<<<g2, b2, 0, stream>>>(Qp, Kp, Vt, km, out);
}
